// TriangleAttentionEndingNode_31645319037644
// MI455X (gfx1250) — matrix operand order repaired and checked by reading, not compiled, not run on hardware
//
#include <hip/hip_runtime.h>

#define NN 320
#define DD 128
#define HH 4
#define CC 32

typedef __attribute__((ext_vector_type(16))) __bf16 v16bf;
typedef __attribute__((ext_vector_type(8))) float v8f;
typedef unsigned int u32x4 __attribute__((ext_vector_type(4)));
typedef int i32x4 __attribute__((ext_vector_type(4)));
typedef int i32x8 __attribute__((ext_vector_type(8)));

#if __has_builtin(__builtin_amdgcn_tensor_load_to_lds)
#define HAVE_TDM 1
#else
#define HAVE_TDM 0
#endif

struct FragRaw { uint4 lo, hi; };
union FragCvt { FragRaw r; v16bf v; };

// Native bf16 convert: backend selects the hardware cvt op instead of 4 VALU ops.
static __device__ __forceinline__ unsigned short f2bf(float f) {
  return __builtin_bit_cast(unsigned short, (__bf16)f);
}
static __device__ __forceinline__ float bf2f(unsigned short h) {
  return __uint_as_float(((unsigned int)h) << 16);
}

// A-fragment 16x32 bf16 (ISA 7.12.2): lane m=lane&15; lanes 0-15 hold K=k0..k0+7
// in VGPR0-3 and K=k0+16..k0+23 in VGPR4-7; lanes 16-31 hold K=+8 / +24.
static __device__ __forceinline__ v16bf load_a_frag(const unsigned short* A, int ld,
                                                    int m0, int k0, int lane) {
  const int m = lane & 15, half = lane >> 4;
  FragCvt c;
  const unsigned short* p = A + (size_t)(m0 + m) * ld + k0 + half * 8;
  c.r.lo = *(const uint4*)(p);
  c.r.hi = *(const uint4*)(p + 16);
  return c.v;
}
// B-fragment 32x16 bf16 from TRANSPOSED storage BT[n][k] (row stride ld):
// lane n=lane&15 holds column n; lanes 0-15 hold K=k0..k0+7 and k0+16..k0+23, lanes 16-31 K=+8 / +24.
static __device__ __forceinline__ v16bf load_b_frag(const unsigned short* BT, int ld,
                                                    int n0, int k0, int lane) {
  const int n = lane & 15, half = lane >> 4;
  FragCvt c;
  const unsigned short* p = BT + (size_t)(n0 + n) * ld + k0 + half * 8;
  c.r.lo = *(const uint4*)(p);
  c.r.hi = *(const uint4*)(p + 16);
  return c.v;
}

static __device__ __forceinline__ v8f wmma_bf16(v16bf a, v16bf b, v8f c) {
  return __builtin_amdgcn_wmma_f32_16x16x32_bf16(false, a, false, b, (short)0, c,
                                                 false, false);
}

// ---------------- Kernel 1: LayerNorm + bf16 convert (one wave per 128-vector)
__global__ __launch_bounds__(256) void ln_kernel(const float* __restrict__ Z,
                                                 const float* __restrict__ gamma,
                                                 const float* __restrict__ beta,
                                                 unsigned short* __restrict__ LNZ) {
  const int lane = threadIdx.x & 31;
  const int wave = threadIdx.x >> 5;
  const size_t vec = (size_t)blockIdx.x * 8 + wave;
  if (vec >= (size_t)NN * NN) return;
  float4 x = *(const float4*)(Z + vec * DD + lane * 4);
  float s = x.x + x.y + x.z + x.w;
  float s2 = x.x * x.x + x.y * x.y + x.z * x.z + x.w * x.w;
#pragma unroll
  for (int m = 1; m < 32; m <<= 1) {
    s += __shfl_xor(s, m, 32);
    s2 += __shfl_xor(s2, m, 32);
  }
  const float mu = s * (1.0f / DD);
  const float var = s2 * (1.0f / DD) - mu * mu;
  const float rstd = rsqrtf(var + 1e-5f);
  float4 g = *(const float4*)(gamma + lane * 4);
  float4 b = *(const float4*)(beta + lane * 4);
  unsigned int p0 = (unsigned)f2bf((x.x - mu) * rstd * g.x + b.x) |
                    ((unsigned)f2bf((x.y - mu) * rstd * g.y + b.y) << 16);
  unsigned int p1 = (unsigned)f2bf((x.z - mu) * rstd * g.z + b.z) |
                    ((unsigned)f2bf((x.w - mu) * rstd * g.w + b.w) << 16);
  *(uint2*)(LNZ + vec * DD + lane * 4) = make_uint2(p0, p1);
}

// ---------------- Kernel 2: transpose all five 128x128 weights to bf16
__global__ void wprep_kernel(const float* __restrict__ wq, const float* __restrict__ wk,
                             const float* __restrict__ wv, const float* __restrict__ wg,
                             const float* __restrict__ wo,
                             unsigned short* __restrict__ W) {
  int idx = blockIdx.x * blockDim.x + threadIdx.x;
  if (idx >= 5 * DD * DD) return;
  const int m = idx / (DD * DD);
  const int rem = idx - m * DD * DD;
  const int c = rem / DD;        // output row = input col
  const int r = rem - c * DD;    // output col = input row
  const float* src = (m == 0) ? wq : (m == 1) ? wk : (m == 2) ? wv : (m == 3) ? wg : wo;
  W[idx] = f2bf(src[r * DD + c]);
}

// ---------------- Kernel 3: tri_bias[h][q][k] = sum_d LNZ[k][q][d] * w_b[d][h]
__global__ __launch_bounds__(256) void bias_kernel(const unsigned short* __restrict__ LNZ,
                                                   const float* __restrict__ wb,
                                                   float* __restrict__ tri) {
  const int lane = threadIdx.x & 31;
  const int wave = threadIdx.x >> 5;
  const size_t p = (size_t)blockIdx.x * 8 + wave;
  if (p >= (size_t)NN * NN) return;
  const int q = (int)(p % NN);   // consecutive waves -> consecutive q -> coalesced LNZ
  const int k = (int)(p / NN);
  uint2 raw = *(const uint2*)(LNZ + ((size_t)k * NN + q) * DD + lane * 4);
  float x0 = bf2f((unsigned short)(raw.x & 0xFFFFu));
  float x1 = bf2f((unsigned short)(raw.x >> 16));
  float x2 = bf2f((unsigned short)(raw.y & 0xFFFFu));
  float x3 = bf2f((unsigned short)(raw.y >> 16));
  const float* w0 = wb + (lane * 4) * HH;
  float acc[HH];
#pragma unroll
  for (int h = 0; h < HH; h++)
    acc[h] = x0 * w0[h] + x1 * w0[HH + h] + x2 * w0[2 * HH + h] + x3 * w0[3 * HH + h];
#pragma unroll
  for (int m = 1; m < 32; m <<= 1) {
#pragma unroll
    for (int h = 0; h < HH; h++) acc[h] += __shfl_xor(acc[h], m, 32);
  }
  if (lane == 0) {
#pragma unroll
    for (int h = 0; h < HH; h++) tri[((size_t)h * NN + q) * NN + k] = acc[h];
  }
}

// ---------------- Kernel 4: fused attention, one workgroup per row r
#define KPAD 328  // padded k-stride for transposed V (16 distinct LDS banks per 16 lanes)

__global__ __launch_bounds__(256) void attn_kernel(
    const unsigned short* __restrict__ LNZ, const float* __restrict__ Zmask,
    const float* __restrict__ tri, const unsigned short* __restrict__ WqT,
    const unsigned short* __restrict__ WkT, const unsigned short* __restrict__ WvT,
    const unsigned short* __restrict__ WgT, const unsigned short* __restrict__ WoT,
    const float* __restrict__ bg, const float* __restrict__ Zraw,
    const float* __restrict__ out_bias, float* __restrict__ out) {
  extern __shared__ unsigned short smem[];
  unsigned short* Xsm = smem;             // [320][128] LN column, bf16 (80 KB)
  unsigned short* Ksm = smem + 40960;     // [320][128] K projection   (80 KB)
  unsigned short* VsmT = smem + 81920;    // [128][KPAD] V transposed  (82 KB)
  unsigned short* wbuf = smem + 123904;   // [8 waves][16*32] staging  (8 KB)

  const int r = blockIdx.x;
  const int tid = threadIdx.x;
  const int lane = tid & 31;
  const int wave = tid >> 5;
  const int n16 = lane & 15;
  const int half = lane >> 4;

  // ---- Phase A: stage LN column r (Xsm[q][d] = LNZ[q][r][d]) via the TDM ----
#if HAVE_TDM
  if (tid < 32) {  // one wave issues the DMA descriptor (TDM ignores EXEC)
    const unsigned long long ga =
        (unsigned long long)LNZ + (unsigned long long)r * (DD * 2);
    // D# group0: count=1 | lds_addr=0 (Xsm base) | global_addr | type=2
    u32x4 g0 = {1u, 0u, (unsigned)ga,
                (unsigned)((ga >> 32) & 0x1FFFFFFu) | 0x80000000u};
    // D# group1: data_size=2B; tensor_dim0=128, tensor_dim1=320;
    //            tile_dim0=128, tile_dim1=320; tensor_dim0_stride=NN*DD=40960
    i32x8 g1 = {0x00010000, 0x00800000, 0x01400000, 0x00800000,
                320, 40960, 0, 0};
    i32x4 gz = {0, 0, 0, 0};
#if __clang_major__ >= 23
    i32x8 gz8 = {0, 0, 0, 0, 0, 0, 0, 0};
    __builtin_amdgcn_tensor_load_to_lds(g0, g1, gz, gz, gz8, 0);
#else
    __builtin_amdgcn_tensor_load_to_lds(g0, g1, gz, gz, 0);
#endif
#if __has_builtin(__builtin_amdgcn_s_wait_tensorcnt)
    __builtin_amdgcn_s_wait_tensorcnt(0);
#else
    asm volatile("s_wait_tensorcnt 0" ::: "memory");
#endif
  }
#else
  for (int i = tid; i < (NN * DD / 8); i += 256) {
    const int row = i >> 4;
    const int c8 = (i & 15) << 3;
    *(uint4*)(Xsm + row * DD + c8) =
        *(const uint4*)(LNZ + ((size_t)row * NN + r) * DD + c8);
  }
#endif
  __syncthreads();

  // Phase B: K and V projections via WMMA (tiles 0..159 -> K, 160..319 -> V)
  for (int t = wave; t < 320; t += 8) {
    const bool isV = t >= 160;
    const int tt = isV ? (t - 160) : t;
    const int mt = tt >> 3, nt = tt & 7;
    const unsigned short* WT = isV ? WvT : WkT;
    v8f acc = {};
#pragma unroll
    for (int ks = 0; ks < 4; ks++) {
      v16bf a = load_a_frag(Xsm, DD, mt * 16, ks * 32, lane);
      v16bf b = load_b_frag(WT, DD, nt * 16, ks * 32, lane);
      acc = wmma_bf16(a, b, acc);
    }
    if (!isV) {
#pragma unroll
      for (int j = 0; j < 8; j++)
        Ksm[(mt * 16 + half * 8 + j) * DD + nt * 16 + n16] = f2bf(acc[j]);
    } else {
#pragma unroll
      for (int j = 0; j < 8; j++)
        VsmT[(nt * 16 + n16) * KPAD + mt * 16 + half * 8 + j] = f2bf(acc[j]);
    }
  }
  __syncthreads();
  // No workgroup barriers past this point (per-wave loop is divergent).

  unsigned short* myb = wbuf + wave * 512;
  const float qscale = 0.17677669529663687f;  // 1/sqrt(32)
  v8f vzero = {};

  for (int qt = wave; qt < 20; qt += 8) {
    const int q0 = qt * 16;
    v8f oacc[8];
#pragma unroll
    for (int dt = 0; dt < 8; dt++) oacc[dt] = vzero;

    for (int h = 0; h < HH; h++) {
      const int ch0 = h * CC;

      // --- Q projection (scaled), staged through LDS into A-frag layout ---
#pragma unroll
      for (int nt = 0; nt < 2; nt++) {
        v8f acc = vzero;
#pragma unroll
        for (int ks = 0; ks < 4; ks++) {
          v16bf a = load_a_frag(Xsm, DD, q0, ks * 32, lane);
          v16bf b = load_b_frag(WqT, DD, ch0 + nt * 16, ks * 32, lane);
          acc = wmma_bf16(a, b, acc);
        }
#pragma unroll
        for (int j = 0; j < 8; j++)
          myb[(half * 8 + j) * 32 + nt * 16 + n16] = f2bf(acc[j] * qscale);
      }
      asm volatile("s_wait_dscnt 0" ::: "memory");
      v16bf qfrag = load_a_frag(myb, 32, 0, 0, lane);

      // --- flash-style online softmax over 320 keys, 32 per step ---
      float mrow[8], lrow[8];
#pragma unroll
      for (int j = 0; j < 8; j++) { mrow[j] = -3.0e38f; lrow[j] = 0.0f; }
      v8f o0 = vzero, o1 = vzero;

      for (int kc = 0; kc < 10; kc++) {
        const int kk0 = kc * 32;
        v16bf kb0 = load_b_frag(Ksm, DD, kk0, ch0, lane);
        v16bf kb1 = load_b_frag(Ksm, DD, kk0 + 16, ch0, lane);
        v8f S0 = wmma_bf16(qfrag, kb0, vzero);
        v8f S1 = wmma_bf16(qfrag, kb1, vzero);

        // add mask bias + triangle bias (C/D layout: row = q0+half*8+j, col = n16)
        {
          const int k0i = kk0 + n16, k1i = kk0 + 16 + n16;
          const float mb0 = 1.0e9f * (Zmask[(size_t)k0i * NN + r] - 1.0f);
          const float mb1 = 1.0e9f * (Zmask[(size_t)k1i * NN + r] - 1.0f);
          const float* tb = tri + ((size_t)h * NN + q0 + half * 8) * NN;
          if (kc + 1 < 10)  // hide tri latency for the next key block
            __builtin_prefetch(tb + kk0 + 32 + n16, 0, 1);
#pragma unroll
          for (int j = 0; j < 8; j++) {
            S0[j] += mb0 + tb[(size_t)j * NN + k0i];
            S1[j] += mb1 + tb[(size_t)j * NN + k1i];
          }
        }
        // row max across the 16 columns (xor masks < 16 stay inside each half)
        float tmax[8];
#pragma unroll
        for (int j = 0; j < 8; j++) tmax[j] = fmaxf(S0[j], S1[j]);
#pragma unroll
        for (int x = 1; x < 16; x <<= 1)
#pragma unroll
          for (int j = 0; j < 8; j++)
            tmax[j] = fmaxf(tmax[j], __shfl_xor(tmax[j], x, 32));

        float rs[8];
#pragma unroll
        for (int j = 0; j < 8; j++) {
          const float nm = fmaxf(mrow[j], tmax[j]);
          const float alpha = __expf(mrow[j] - nm);
          mrow[j] = nm;
          lrow[j] *= alpha;
          o0[j] *= alpha;
          o1[j] *= alpha;
          S0[j] = __expf(S0[j] - nm);
          S1[j] = __expf(S1[j] - nm);
          rs[j] = S0[j] + S1[j];
        }
#pragma unroll
        for (int x = 1; x < 16; x <<= 1)
#pragma unroll
          for (int j = 0; j < 8; j++) rs[j] += __shfl_xor(rs[j], x, 32);
#pragma unroll
        for (int j = 0; j < 8; j++) lrow[j] += rs[j];

        // P -> bf16 -> LDS -> A-frag; O += P @ V (V read from transposed LDS)
#pragma unroll
        for (int j = 0; j < 8; j++) {
          myb[(half * 8 + j) * 32 + n16] = f2bf(S0[j]);
          myb[(half * 8 + j) * 32 + 16 + n16] = f2bf(S1[j]);
        }
        asm volatile("s_wait_dscnt 0" ::: "memory");
        v16bf pfrag = load_a_frag(myb, 32, 0, 0, lane);
        v16bf vb0 = load_b_frag(VsmT, KPAD, ch0, kk0, lane);
        v16bf vb1 = load_b_frag(VsmT, KPAD, ch0 + 16, kk0, lane);
        o0 = wmma_bf16(pfrag, vb0, o0);
        o1 = wmma_bf16(pfrag, vb1, o1);
      }

      // normalize
#pragma unroll
      for (int j = 0; j < 8; j++) {
        const float inv = 1.0f / lrow[j];
        o0[j] *= inv;
        o1[j] *= inv;
      }

      // --- gate: sigmoid(X@wg + bg) ---
#pragma unroll
      for (int nt = 0; nt < 2; nt++) {
        v8f gacc = vzero;
#pragma unroll
        for (int ks = 0; ks < 4; ks++) {
          v16bf a = load_a_frag(Xsm, DD, q0, ks * 32, lane);
          v16bf b = load_b_frag(WgT, DD, ch0 + nt * 16, ks * 32, lane);
          gacc = wmma_bf16(a, b, gacc);
        }
        const float bgv = bg[ch0 + nt * 16 + n16];
#pragma unroll
        for (int j = 0; j < 8; j++) {
          const float g = 1.0f / (1.0f + __expf(-(gacc[j] + bgv)));
          if (nt == 0) o0[j] *= g; else o1[j] *= g;
        }
      }

      // --- output projection: accumulate (O*G)[16x32] @ wo[ch0:ch0+32][0:128] ---
#pragma unroll
      for (int j = 0; j < 8; j++) {
        myb[(half * 8 + j) * 32 + n16] = f2bf(o0[j]);
        myb[(half * 8 + j) * 32 + 16 + n16] = f2bf(o1[j]);
      }
      asm volatile("s_wait_dscnt 0" ::: "memory");
      v16bf ogfrag = load_a_frag(myb, 32, 0, 0, lane);
#pragma unroll
      for (int dt = 0; dt < 8; dt++) {
        v16bf wb = load_b_frag(WoT, DD, dt * 16, ch0, lane);
        oacc[dt] = wmma_bf16(ogfrag, wb, oacc[dt]);
      }
    }  // heads

    // epilogue: result[q][r][d] = Z_raw[q][r][d] + out + out_bias
#pragma unroll
    for (int dt = 0; dt < 8; dt++) {
      const int d = dt * 16 + n16;
      const float ob = out_bias[d];
#pragma unroll
      for (int j = 0; j < 8; j++) {
        const int q = q0 + half * 8 + j;
        const size_t idx = ((size_t)q * NN + r) * DD + d;
        out[idx] = Zraw[idx] + oacc[dt][j] + ob;
      }
    }
  }
}

extern "C" void kernel_launch(void* const* d_in, const int* in_sizes, int n_in,
                              void* d_out, int out_size, void* d_ws, size_t ws_size,
                              hipStream_t stream) {
  const float* Zraw = (const float*)d_in[0];
  const float* Zmask = (const float*)d_in[1];
  const float* ln_g = (const float*)d_in[2];
  const float* ln_b = (const float*)d_in[3];
  const float* w_b = (const float*)d_in[4];
  const float* wq = (const float*)d_in[5];
  const float* wk = (const float*)d_in[6];
  const float* wv = (const float*)d_in[7];
  const float* wg = (const float*)d_in[8];
  const float* bg = (const float*)d_in[9];
  const float* wo = (const float*)d_in[10];
  const float* outb = (const float*)d_in[11];
  float* out = (float*)d_out;

  char* ws = (char*)d_ws;
  unsigned short* LNZ = (unsigned short*)ws;                       // 26,214,400 B
  unsigned short* Wt = (unsigned short*)(ws + 26214400);           //    163,840 B
  float* tri = (float*)(ws + 26214400 + 163840);                   //  1,638,400 B

  unsigned short* WqT = Wt;
  unsigned short* WkT = Wt + 16384;
  unsigned short* WvT = Wt + 32768;
  unsigned short* WgT = Wt + 49152;
  unsigned short* WoT = Wt + 65536;

  ln_kernel<<<12800, 256, 0, stream>>>(Zraw, ln_g, ln_b, LNZ);
  wprep_kernel<<<(5 * DD * DD + 255) / 256, 256, 0, stream>>>(wq, wk, wv, wg, wo, Wt);
  bias_kernel<<<12800, 256, 0, stream>>>(LNZ, w_b, tri);

  const int smem_bytes = 256000;  // 250 KB of the 320 KB/WGP LDS
  hipFuncSetAttribute(reinterpret_cast<const void*>(attn_kernel),
                      hipFuncAttributeMaxDynamicSharedMemorySize, smem_bytes);
  attn_kernel<<<NN, 256, smem_bytes, stream>>>(LNZ, Zmask, tri, WqT, WkT, WvT, WgT,
                                               WoT, bg, Zraw, outb, out);
}
